// TimeGNN_66425964200488
// MI455X (gfx1250) — compile-verified, matrix operand order checked by reading, not run on hardware
//
#include <hip/hip_runtime.h>
#include <hip/hip_bf16.h>

// ---------------------------------------------------------------------------
// TimeGNN forward for MI455X (gfx1250, wave32, WMMA bf16 16x16x32, f32 accum)
// B=8, S=256, F=8, H=64, OUT=1
// ---------------------------------------------------------------------------

typedef __attribute__((ext_vector_type(16))) __bf16 v16bf;
typedef __attribute__((ext_vector_type(8)))  __bf16 v8bf;
typedef __attribute__((ext_vector_type(8)))  float  v8f;

#define Bn 8
#define Sn 256
#define Fn 8
#define Hn 64
#define ROWS (Bn*Sn)          // 2048
#define EPSV 1e-5f

// ---- workspace layout (bytes, all 256-aligned) ----
#define WS_HC_BF   0u          // 2048*64 bf16        (x / hc, row-major)
#define WS_HC_BFT  262144u     // (B,H,S) bf16        (hc transposed)
#define WS_CAT_BF  524288u     // 2048*192 bf16       (conv concat)
#define WS_U       1310720u    // 2048*64 f32
#define WS_V       1835008u    // 2048*64 f32
#define WS_ADJ     2359296u    // (B,j,i) bf16  adj^T
#define WS_DEGINV  3407872u    // 2048 f32
#define WS_AGG_BF  3416064u    // 2048*64 bf16
#define WS_HC_PRE  3678208u    // 2048*64 f32 (pre-BN)
#define WS_STATS   4202496u    // 128 f32 (sum, sumsq)
#define WS_XLACC   4203008u    // 512 f32
#define WS_E1U     4205056u    // 64x64 bf16 (e1w top, transposed)
#define WS_E1V     4213248u    // 64x64 bf16 (e1w bot, transposed)
#define WS_WLT     4221440u    // 3x 64x64 bf16
#define WS_WRT     4246016u    // 3x 64x64 bf16
#define WS_FUSWT   4270592u    // 64x192 bf16 (fus_w transposed)

static __device__ __forceinline__ v8f zero8() {
  v8f z;
#pragma unroll
  for (int i = 0; i < 8; ++i) z[i] = 0.f;
  return z;
}

// Load one 16x32 bf16 fragment (A-layout; B uses the same pattern on the
// transposed weight matrix). base -> element [0,0] of the 16-row tile.
// Lane holds row (lane&15); K runs [k0+8*half, +7] and [k0+16+8*half, +7].
static __device__ __forceinline__ v16bf load_frag(const __bf16* __restrict__ base,
                                                  int ld, int k0, int lane) {
  const int r = lane & 15, half = lane >> 4;
  const __bf16* p = base + r * ld + k0 + 8 * half;
  v8bf lo = *reinterpret_cast<const v8bf*>(p);
  v8bf hi = *reinterpret_cast<const v8bf*>(p + 16);
  v16bf out;
#pragma unroll
  for (int t = 0; t < 8; ++t) { out[t] = lo[t]; out[t + 8] = hi[t]; }
  return out;
}

static __device__ __forceinline__ v8f wmma_bf16(v16bf a, v16bf b, v8f c) {
  return __builtin_amdgcn_wmma_f32_16x16x32_bf16(false, a, false, b,
                                                 (short)0, c, false, false);
}

// ---------------------------------------------------------------------------
// K0: transpose/convert weights to bf16 WMMA-B layout; zero xl accumulator
// ---------------------------------------------------------------------------
__global__ void k0_prep(const float* __restrict__ e1w, const float* __restrict__ fusw,
                        const float* w0l, const float* w0r,
                        const float* w1l, const float* w1r,
                        const float* w2l, const float* w2r,
                        __bf16* e1u, __bf16* e1v, __bf16* wlt, __bf16* wrt,
                        __bf16* fuswT, float* xl_acc) {
  int idx = blockIdx.x * blockDim.x + threadIdx.x;
  if (idx < Bn * Hn) xl_acc[idx] = 0.f;
  if (idx < 8 * 4096) {
    int mat = idx >> 12, r = idx & 4095;
    int n = r >> 6, k = r & 63;
    float v = 0.f; __bf16* dst = e1u;
    switch (mat) {
      case 0: v = e1w[k * 64 + n];        dst = e1u;        break;
      case 1: v = e1w[(k + 64) * 64 + n]; dst = e1v;        break;
      case 2: v = w0l[k * 64 + n];        dst = wlt;        break;
      case 3: v = w0r[k * 64 + n];        dst = wrt;        break;
      case 4: v = w1l[k * 64 + n];        dst = wlt + 4096; break;
      case 5: v = w1r[k * 64 + n];        dst = wrt + 4096; break;
      case 6: v = w2l[k * 64 + n];        dst = wlt + 8192; break;
      case 7: v = w2r[k * 64 + n];        dst = wrt + 8192; break;
    }
    dst[n * 64 + k] = (__bf16)v;
  } else if (idx < 8 * 4096 + 64 * 192) {
    int r = idx - 8 * 4096;           // fuswT[n*192+k] = fusw[k*64+n]
    int n = r / 192, k = r - n * 192;
    fuswT[r] = (__bf16)fusw[k * 64 + n];
  }
}

// ---------------------------------------------------------------------------
// K1: conv branches (pointwise + dilated conv, SAME zero-pad) -> cat bf16
// one block per batch, thread = sequence position
// ---------------------------------------------------------------------------
__global__ void k1_features(const float* __restrict__ data,
                            const float* c1w1, const float* c1b1,
                            const float* c1w2, const float* c1b2,
                            const float* c2w1, const float* c2b1,
                            const float* c2w2, const float* c2b2,
                            const float* c3w1, const float* c3b1,
                            __bf16* __restrict__ cat_bf) {
  __shared__ float p[64 * 256];          // [channel][s] : 64 KB LDS
  const int b = blockIdx.x, s = threadIdx.x;
  float df[8];
#pragma unroll
  for (int f = 0; f < 8; ++f) df[f] = data[(b * Sn + s) * Fn + f];
  __bf16* catp = cat_bf + (size_t)(b * Sn + s) * 192;

  // ---- branch 1: pointwise -> relu -> conv(k=3, dil=3) -> relu ----
#pragma unroll
  for (int c = 0; c < 64; ++c) {
    float t = c1b1[c];
#pragma unroll
    for (int f = 0; f < 8; ++f) t = fmaf(c1w1[c * 8 + f], df[f], t);
    p[c * 256 + s] = fmaxf(t, 0.f);
  }
  __syncthreads();
  {
    float xacc[64];
#pragma unroll
    for (int c = 0; c < 64; ++c) xacc[c] = 0.f;
    for (int cp = 0; cp < 64; ++cp) {
      float t0 = (s >= 3)   ? p[cp * 256 + s - 3] : 0.f;
      float t1 =              p[cp * 256 + s];
      float t2 = (s <= 252) ? p[cp * 256 + s + 3] : 0.f;
      const float* w = c1w2 + cp * 3;    // [c,cp,k] = c*192 + cp*3 + k
#pragma unroll
      for (int c = 0; c < 64; ++c) {
        float a = xacc[c];
        a = fmaf(w[c * 192 + 0], t0, a);
        a = fmaf(w[c * 192 + 1], t1, a);
        a = fmaf(w[c * 192 + 2], t2, a);
        xacc[c] = a;
      }
    }
#pragma unroll
    for (int c = 0; c < 64; ++c)
      catp[c] = (__bf16)fmaxf(xacc[c] + c1b2[c], 0.f);
  }
  __syncthreads();

  // ---- branch 2: pointwise -> relu -> conv(k=5, dil=5) -> relu ----
#pragma unroll
  for (int c = 0; c < 64; ++c) {
    float t = c2b1[c];
#pragma unroll
    for (int f = 0; f < 8; ++f) t = fmaf(c2w1[c * 8 + f], df[f], t);
    p[c * 256 + s] = fmaxf(t, 0.f);
  }
  __syncthreads();
  {
    float xacc[64];
#pragma unroll
    for (int c = 0; c < 64; ++c) xacc[c] = 0.f;
    for (int cp = 0; cp < 64; ++cp) {
      float t0 = (s >= 10)  ? p[cp * 256 + s - 10] : 0.f;
      float t1 = (s >= 5)   ? p[cp * 256 + s - 5]  : 0.f;
      float t2 =              p[cp * 256 + s];
      float t3 = (s <= 250) ? p[cp * 256 + s + 5]  : 0.f;
      float t4 = (s <= 245) ? p[cp * 256 + s + 10] : 0.f;
      const float* w = c2w2 + cp * 5;    // [c,cp,k] = c*320 + cp*5 + k
#pragma unroll
      for (int c = 0; c < 64; ++c) {
        float a = xacc[c];
        a = fmaf(w[c * 320 + 0], t0, a);
        a = fmaf(w[c * 320 + 1], t1, a);
        a = fmaf(w[c * 320 + 2], t2, a);
        a = fmaf(w[c * 320 + 3], t3, a);
        a = fmaf(w[c * 320 + 4], t4, a);
        xacc[c] = a;
      }
    }
#pragma unroll
    for (int c = 0; c < 64; ++c)
      catp[64 + c] = (__bf16)fmaxf(xacc[c] + c2b2[c], 0.f);
  }

  // ---- branch 3: pointwise -> relu ----
#pragma unroll
  for (int c = 0; c < 64; ++c) {
    float t = c3b1[c];
#pragma unroll
    for (int f = 0; f < 8; ++f) t = fmaf(c3w1[c * 8 + f], df[f], t);
    catp[128 + c] = (__bf16)fmaxf(t, 0.f);
  }
}

// ---------------------------------------------------------------------------
// K2: fusion GEMM  x = relu(cat(2048,192) @ fus_w(192,64) + b)  [WMMA]
// one wave per 16-row tile; writes hc_bf (row-major) and hc_bfT (B,H,S)
// ---------------------------------------------------------------------------
__global__ void k2_fusion(const __bf16* __restrict__ cat_bf,
                          const __bf16* __restrict__ fuswT,
                          const float* __restrict__ fusb,
                          __bf16* __restrict__ hc_bf, __bf16* __restrict__ hc_bfT) {
  const int lane = threadIdx.x;
  const int row0 = blockIdx.x * 16;
  v8f acc[4];
#pragma unroll
  for (int nt = 0; nt < 4; ++nt) acc[nt] = zero8();
  const __bf16* A = cat_bf + (size_t)row0 * 192;
#pragma unroll
  for (int ks = 0; ks < 6; ++ks) {
    v16bf a = load_frag(A, 192, ks * 32, lane);
#pragma unroll
    for (int nt = 0; nt < 4; ++nt) {
      v16bf bw = load_frag(fuswT + nt * 16 * 192, 192, ks * 32, lane);
      acc[nt] = wmma_bf16(a, bw, acc[nt]);
    }
  }
  const int half = lane >> 4, col = lane & 15;
#pragma unroll
  for (int nt = 0; nt < 4; ++nt) {
    float bb = fusb[nt * 16 + col];
#pragma unroll
    for (int r = 0; r < 8; ++r) {
      int row = row0 + r + 8 * half;
      float xv = fmaxf(acc[nt][r] + bb, 0.f);
      int c = nt * 16 + col;
      hc_bf[row * 64 + c] = (__bf16)xv;
      int b = row >> 8, ss = row & 255;
      hc_bfT[b * (Hn * Sn) + c * Sn + ss] = (__bf16)xv;
    }
  }
}

// ---------------------------------------------------------------------------
// K3: edge-MLP factorization  u = x@e1w[:H],  v = x@e1w[H:]   [WMMA]
// ---------------------------------------------------------------------------
__global__ void k3_uv(const __bf16* __restrict__ x_bf,
                      const __bf16* __restrict__ e1u, const __bf16* __restrict__ e1v,
                      float* __restrict__ u, float* __restrict__ v) {
  const int lane = threadIdx.x;
  const int row0 = blockIdx.x * 16;
  v8f ua[4], va[4];
#pragma unroll
  for (int nt = 0; nt < 4; ++nt) { ua[nt] = zero8(); va[nt] = zero8(); }
  const __bf16* A = x_bf + (size_t)row0 * 64;
#pragma unroll
  for (int ks = 0; ks < 2; ++ks) {
    v16bf a = load_frag(A, 64, ks * 32, lane);
#pragma unroll
    for (int nt = 0; nt < 4; ++nt) {
      v16bf bu = load_frag(e1u + nt * 16 * 64, 64, ks * 32, lane);
      ua[nt] = wmma_bf16(a, bu, ua[nt]);
      v16bf bv = load_frag(e1v + nt * 16 * 64, 64, ks * 32, lane);
      va[nt] = wmma_bf16(a, bv, va[nt]);
    }
  }
  const int half = lane >> 4, col = lane & 15;
#pragma unroll
  for (int nt = 0; nt < 4; ++nt)
#pragma unroll
    for (int r = 0; r < 8; ++r) {
      int row = row0 + r + 8 * half;
      u[row * 64 + nt * 16 + col] = ua[nt][r];
      v[row * 64 + nt * 16 + col] = va[nt][r];
    }
}

// ---------------------------------------------------------------------------
// K4: adjacency: adj[b,i,j] = 1{i>j and l0>=l1}, stored transposed [b][j][i];
//     deg_inv[b,j] = 1/max(1, sum_i adj).  One block per (b,j), thread = i.
// ---------------------------------------------------------------------------
__global__ void k4_adj(const float* __restrict__ u, const float* __restrict__ v,
                       const float* __restrict__ e1b, const float* __restrict__ e2w,
                       const float* __restrict__ e2b, const float* __restrict__ gumbel,
                       __bf16* __restrict__ adj_bf, float* __restrict__ deg_inv) {
  __shared__ float vj[64], w0[64], w1[64];
  __shared__ float red[256];
  const int b = blockIdx.x >> 8, j = blockIdx.x & 255, i = threadIdx.x;
  if (i < 64)        vj[i]       = v[(b * Sn + j) * 64 + i] + e1b[i];
  else if (i < 128)  w0[i - 64]  = e2w[2 * (i - 64)];
  else if (i < 192)  w1[i - 128] = e2w[2 * (i - 128) + 1];
  __syncthreads();

  float ur[64];
  const float4* up = (const float4*)(u + (size_t)(b * Sn + i) * 64);
#pragma unroll
  for (int m = 0; m < 16; ++m) {
    float4 q = up[m];
    ur[4 * m] = q.x; ur[4 * m + 1] = q.y; ur[4 * m + 2] = q.z; ur[4 * m + 3] = q.w;
  }
  float s0 = e2b[0], s1 = e2b[1];
#pragma unroll
  for (int c = 0; c < 64; ++c) {
    float a = fmaxf(ur[c] + vj[c], 0.f);
    s0 = fmaf(a, w0[c], s0);
    s1 = fmaf(a, w1[c], s1);
  }
  const float* g = gumbel + ((size_t)b * (Sn * Sn) + (size_t)i * Sn + j) * 2;
  float l0 = s0 + g[0], l1 = s1 + g[1];          // /TAU preserves argmax
  float adjv = (i > j && l0 >= l1) ? 1.0f : 0.0f;
  adj_bf[(size_t)b * (Sn * Sn) + (size_t)j * Sn + i] = (__bf16)adjv;

  red[i] = adjv;
  __syncthreads();
  for (int off = 128; off > 0; off >>= 1) {
    if (i < off) red[i] += red[i + off];
    __syncthreads();
  }
  if (i == 0) deg_inv[b * Sn + j] = 1.0f / fmaxf(red[0], 1.0f);
}

// ---------------------------------------------------------------------------
// K5a: agg[b,j,h] = deg_inv[b,j] * sum_i adj[b,i,j] * hc[b,i,h]   [WMMA]
//      A = adj^T[b] (j,i), B via hc_bfT (h,i). Also zeroes BN stats.
// ---------------------------------------------------------------------------
__global__ void k5a_agg(const __bf16* __restrict__ adj_bf,
                        const __bf16* __restrict__ hc_bfT,
                        const float* __restrict__ deg_inv,
                        __bf16* __restrict__ agg_bf, float* __restrict__ stats) {
  const int lane = threadIdx.x;
  if (blockIdx.x == 0)
    for (int t = lane; t < 128; t += 32) stats[t] = 0.f;
  const int b = blockIdx.x >> 4, j0 = (blockIdx.x & 15) * 16;
  v8f acc[4];
#pragma unroll
  for (int nt = 0; nt < 4; ++nt) acc[nt] = zero8();
  const __bf16* A  = adj_bf + (size_t)b * (Sn * Sn) + (size_t)j0 * Sn;
  const __bf16* Bt = hc_bfT + b * (Hn * Sn);
#pragma unroll
  for (int ks = 0; ks < 8; ++ks) {
    v16bf a = load_frag(A, Sn, ks * 32, lane);
#pragma unroll
    for (int nt = 0; nt < 4; ++nt) {
      v16bf bw = load_frag(Bt + nt * 16 * Sn, Sn, ks * 32, lane);
      acc[nt] = wmma_bf16(a, bw, acc[nt]);
    }
  }
  const int half = lane >> 4, col = lane & 15;
#pragma unroll
  for (int nt = 0; nt < 4; ++nt)
#pragma unroll
    for (int r = 0; r < 8; ++r) {
      int j = j0 + r + 8 * half;
      float dv = deg_inv[b * Sn + j];
      agg_bf[(size_t)(b * Sn + j) * 64 + nt * 16 + col] = (__bf16)(acc[nt][r] * dv);
    }
}

// ---------------------------------------------------------------------------
// K5b: hc_pre = agg @ wl + hc @ wr + bl   [WMMA] + BN partial sums (atomics)
// ---------------------------------------------------------------------------
__global__ void k5b_linear(const __bf16* __restrict__ agg_bf,
                           const __bf16* __restrict__ hc_bf,
                           const __bf16* __restrict__ wlT, const __bf16* __restrict__ wrT,
                           const float* __restrict__ bl,
                           float* __restrict__ hc_pre, float* __restrict__ stats) {
  const int lane = threadIdx.x;
  const int row0 = blockIdx.x * 16;
  const __bf16* A1 = agg_bf + (size_t)row0 * 64;
  const __bf16* A2 = hc_bf  + (size_t)row0 * 64;
  const int half = lane >> 4, col = lane & 15;
#pragma unroll
  for (int nt = 0; nt < 4; ++nt) {
    v8f acc = zero8();
#pragma unroll
    for (int ks = 0; ks < 2; ++ks) {
      v16bf a1 = load_frag(A1, 64, ks * 32, lane);
      v16bf b1 = load_frag(wlT + nt * 16 * 64, 64, ks * 32, lane);
      acc = wmma_bf16(a1, b1, acc);
      v16bf a2 = load_frag(A2, 64, ks * 32, lane);
      v16bf b2 = load_frag(wrT + nt * 16 * 64, 64, ks * 32, lane);
      acc = wmma_bf16(a2, b2, acc);
    }
    float bb = bl[nt * 16 + col];
    float s1 = 0.f, s2 = 0.f;
#pragma unroll
    for (int r = 0; r < 8; ++r) {
      float vv = acc[r] + bb;
      hc_pre[(row0 + r + 8 * half) * 64 + nt * 16 + col] = vv;
      s1 += vv;
      s2 = fmaf(vv, vv, s2);
    }
    s1 += __shfl_xor(s1, 16);
    s2 += __shfl_xor(s2, 16);
    if (lane < 16) {
      atomicAdd(&stats[nt * 16 + lane], s1);
      atomicAdd(&stats[64 + nt * 16 + lane], s2);
    }
  }
}

// ---------------------------------------------------------------------------
// K5c: apply BN; refresh hc_bf / hc_bfT; accumulate xl at s = S-1
// ---------------------------------------------------------------------------
__global__ void k5c_bn(const float* __restrict__ hc_pre, const float* __restrict__ stats,
                       const float* __restrict__ bng, const float* __restrict__ bnb,
                       const float* __restrict__ gnnw_w, int layer,
                       __bf16* __restrict__ hc_bf, __bf16* __restrict__ hc_bfT,
                       float* __restrict__ xl_acc) {
  const int idx = blockIdx.x * 256 + threadIdx.x;     // < 2048*64
  const int row = idx >> 6, c = idx & 63;
  const float invN = 1.0f / (float)ROWS;
  float mean = stats[c] * invN;
  float var  = stats[64 + c] * invN - mean * mean;
  float scale = bng[c] * rsqrtf(var + EPSV);
  float shift = bnb[c] - mean * scale;
  float hv = fmaf(hc_pre[idx], scale, shift);
  hc_bf[idx] = (__bf16)hv;
  const int b = row >> 8, s = row & 255;
  hc_bfT[b * (Hn * Sn) + c * Sn + s] = (__bf16)hv;
  if (s == Sn - 1)
    xl_acc[b * 64 + c] += gnnw_w[layer] * fmaxf(hv, 0.f);
}

// ---------------------------------------------------------------------------
// K6: head  y = relu(BN_batch(xl@o1w + o1b)) @ o2w + o2b    (tiny, 1 block)
// ---------------------------------------------------------------------------
__global__ void k6_head(const float* __restrict__ xl_acc, const float* __restrict__ gnnw_b,
                        const float* __restrict__ o1w, const float* __restrict__ o1b,
                        const float* __restrict__ obn_g, const float* __restrict__ obn_b,
                        const float* __restrict__ o2w, const float* __restrict__ o2b,
                        float* __restrict__ out) {
  __shared__ float ys[8][32];
  __shared__ float scl[32], sft[32];
  const int t = threadIdx.x, b = t >> 5, o = t & 31;
  float acc = o1b[o];
  float gb = gnnw_b[0];
#pragma unroll
  for (int h = 0; h < 64; ++h)
    acc = fmaf(xl_acc[b * 64 + h] + gb, o1w[h * 32 + o], acc);
  ys[b][o] = acc;
  __syncthreads();
  if (t < 32) {
    float m = 0.f;
#pragma unroll
    for (int bb = 0; bb < 8; ++bb) m += ys[bb][t];
    m *= 0.125f;
    float vv = 0.f;
#pragma unroll
    for (int bb = 0; bb < 8; ++bb) { float d = ys[bb][t] - m; vv = fmaf(d, d, vv); }
    vv *= 0.125f;
    float sc = obn_g[t] * rsqrtf(vv + EPSV);
    scl[t] = sc;
    sft[t] = obn_b[t] - m * sc;
  }
  __syncthreads();
  float val = fmaxf(fmaf(ys[b][o], scl[o], sft[o]), 0.f) * o2w[o];
#pragma unroll
  for (int off = 16; off > 0; off >>= 1) val += __shfl_down(val, off);
  if (o == 0) out[b] = val + o2b[0];
}

// ---------------------------------------------------------------------------
extern "C" void kernel_launch(void* const* d_in, const int* in_sizes, int n_in,
                              void* d_out, int out_size, void* d_ws, size_t ws_size,
                              hipStream_t stream) {
  (void)in_sizes; (void)n_in; (void)out_size; (void)ws_size;
  const float* data   = (const float*)d_in[0];
  const float* gumbel = (const float*)d_in[1];
  const float* c1w1 = (const float*)d_in[2];  const float* c1b1 = (const float*)d_in[3];
  const float* c1w2 = (const float*)d_in[4];  const float* c1b2 = (const float*)d_in[5];
  const float* c2w1 = (const float*)d_in[6];  const float* c2b1 = (const float*)d_in[7];
  const float* c2w2 = (const float*)d_in[8];  const float* c2b2 = (const float*)d_in[9];
  const float* c3w1 = (const float*)d_in[10]; const float* c3b1 = (const float*)d_in[11];
  const float* fusw = (const float*)d_in[12]; const float* fusb = (const float*)d_in[13];
  const float* e1w  = (const float*)d_in[14]; const float* e1b  = (const float*)d_in[15];
  const float* e2w  = (const float*)d_in[16]; const float* e2b  = (const float*)d_in[17];
  const float* gnnw_w = (const float*)d_in[18]; const float* gnnw_b = (const float*)d_in[19];
  const float* o1w  = (const float*)d_in[20]; const float* o1b  = (const float*)d_in[21];
  const float* obng = (const float*)d_in[22]; const float* obnb = (const float*)d_in[23];
  const float* o2w  = (const float*)d_in[24]; const float* o2b  = (const float*)d_in[25];

  char* ws = (char*)d_ws;
  __bf16* hc_bf  = (__bf16*)(ws + WS_HC_BF);
  __bf16* hc_bfT = (__bf16*)(ws + WS_HC_BFT);
  __bf16* cat_bf = (__bf16*)(ws + WS_CAT_BF);
  float*  u      = (float*)(ws + WS_U);
  float*  v      = (float*)(ws + WS_V);
  __bf16* adj_bf = (__bf16*)(ws + WS_ADJ);
  float*  deginv = (float*)(ws + WS_DEGINV);
  __bf16* agg_bf = (__bf16*)(ws + WS_AGG_BF);
  float*  hc_pre = (float*)(ws + WS_HC_PRE);
  float*  stats  = (float*)(ws + WS_STATS);
  float*  xl_acc = (float*)(ws + WS_XLACC);
  __bf16* e1u    = (__bf16*)(ws + WS_E1U);
  __bf16* e1v    = (__bf16*)(ws + WS_E1V);
  __bf16* wlt    = (__bf16*)(ws + WS_WLT);
  __bf16* wrt    = (__bf16*)(ws + WS_WRT);
  __bf16* fuswT  = (__bf16*)(ws + WS_FUSWT);
  float*  out    = (float*)d_out;

  k0_prep<<<176, 256, 0, stream>>>(e1w, fusw,
      (const float*)d_in[26], (const float*)d_in[28],
      (const float*)d_in[31], (const float*)d_in[33],
      (const float*)d_in[36], (const float*)d_in[38],
      e1u, e1v, wlt, wrt, fuswT, xl_acc);

  k1_features<<<Bn, Sn, 0, stream>>>(data, c1w1, c1b1, c1w2, c1b2,
                                     c2w1, c2b1, c2w2, c2b2, c3w1, c3b1, cat_bf);

  k2_fusion<<<ROWS / 16, 32, 0, stream>>>(cat_bf, fuswT, fusb, hc_bf, hc_bfT);

  k3_uv<<<ROWS / 16, 32, 0, stream>>>(hc_bf, e1u, e1v, u, v);

  k4_adj<<<Bn * Sn, 256, 0, stream>>>(u, v, e1b, e2w, e2b, gumbel, adj_bf, deginv);

  for (int l = 0; l < 3; ++l) {
    const float* bl  = (const float*)d_in[27 + 5 * l];
    const float* bng = (const float*)d_in[29 + 5 * l];
    const float* bnb = (const float*)d_in[30 + 5 * l];
    k5a_agg<<<Bn * (Sn / 16), 32, 0, stream>>>(adj_bf, hc_bfT, deginv, agg_bf, stats);
    k5b_linear<<<ROWS / 16, 32, 0, stream>>>(agg_bf, hc_bf, wlt + l * 4096,
                                             wrt + l * 4096, bl, hc_pre, stats);
    k5c_bn<<<(ROWS * Hn) / 256, 256, 0, stream>>>(hc_pre, stats, bng, bnb, gnnw_w, l,
                                                  hc_bf, hc_bfT, xl_acc);
  }

  k6_head<<<1, 256, 0, stream>>>(xl_acc, gnnw_b, o1w, o1b, obng, obnb, o2w, o2b, out);
}
